// EditModel_89120571392208
// MI455X (gfx1250) — matrix operand order repaired and checked by reading, not compiled, not run on hardware
//
#include <hip/hip_runtime.h>
#include <cmath>

typedef __attribute__((ext_vector_type(16))) _Float16 v16h;
typedef __attribute__((ext_vector_type(8)))  float    v8f;
typedef __attribute__((ext_vector_type(4)))  unsigned int v4u;
typedef __attribute__((ext_vector_type(8)))  unsigned int v8u;

namespace {
constexpr int kX2 = 26, kX1 = 25, kCB = 192, kD = 128;
constexpr int kC = 24, kY1 = 25;
constexpr int kH = 256, kHD = 128;
constexpr int kNHead = 128;                       // packed head width: 62 sub + 2 pad + 62 ins + 2 pad
constexpr long kRows = 25L * 25 * 24 * 8;         // 120000 output rows per head
constexpr long kOutHead = kRows * 62;             // 7,440,000
}

// ---------------- WMMA helpers (CDNA5 f16 16x16x32, f32 accum) ----------------

__device__ __forceinline__ v8f wmma_f16(v16h a, v16h b, v8f c) {
  return __builtin_amdgcn_wmma_f32_16x16x32_f16(false, a, false, b, (short)0, c, false, false);
}

// A fragment (16x32, row-major source): row = lane&15, elems 0..7 -> K = kc + hi8 + i,
// elems 8..15 -> K = kc + 16 + hi8 + i  (hi8 = (lane>>4)*8).
template <typename PT>
__device__ __forceinline__ v16h ldA(PT rowbase, int kc, int hi8) {
  v16h a;
  const int o0 = kc + hi8;
  const int o1 = kc + 16 + hi8;
#pragma unroll
  for (int i = 0; i < 8; ++i) { a[i] = rowbase[o0 + i]; a[8 + i] = rowbase[o1 + i]; }
  return a;
}

// B fragment (32x16) from weight stored as (N rows x K cols) row-major (i.e. B = W^T):
// col n = lane&15 (+n0), elems 0..7 -> K = kc + (lane>>4)*8 + i, elems 8..15 -> K = kc + 16 + (lane>>4)*8 + i.
__device__ __forceinline__ v16h ldB(const _Float16* __restrict__ W, int gn, int K, int kc, int kh) {
  const _Float16* p = W + (size_t)gn * K + kc + kh / 2;
  v16h b;
#pragma unroll
  for (int i = 0; i < 8; ++i) { b[i] = p[i]; b[8 + i] = p[16 + i]; }
  return b;
}

__device__ __forceinline__ float sigm(float x) { return 1.0f / (1.0f + __expf(-x)); }

// ---------------- Tensor Data Mover: 1-D contiguous f32 tile, global -> LDS -------------
// D# per CDNA5 ISA (§8.3 group0, §8.4 group1). count=1, type=2 (image),
// data_size=4B, tensor/tile = 1-D run of nelem f32. Groups 2/3 omitted (<=2D tensor).

__device__ __forceinline__ void tdm_load_1d_f32(const float* gsrc, unsigned lds_off_bytes,
                                                unsigned nelem) {
  unsigned long long ga = (unsigned long long)(uintptr_t)gsrc;
  v4u g0;
  g0[0] = 1u;                                               // count=1 (valid), user mode
  g0[1] = lds_off_bytes;                                    // lds_addr (bytes)
  g0[2] = (unsigned)ga;                                     // global_addr[31:0]
  g0[3] = (unsigned)((ga >> 32) & 0x01FFFFFFu) | (2u << 30);// global_addr[56:32] | type=2
  v8u g1;
  g1[0] = 2u << 16;                                         // data_size=2 (4 bytes), no flags
  g1[1] = (nelem & 0xFFFFu) << 16;                          // tensor_dim0[15:0]
  g1[2] = (nelem >> 16) | (1u << 16);                       // tensor_dim0[31:16], tensor_dim1=1
  g1[3] = (nelem & 0xFFFFu) << 16;                          // tensor_dim1[31:16]=0, tile_dim0
  g1[4] = 0u;                                               // tile_dim1=0, tile_dim2=0
  g1[5] = nelem;                                            // tensor_dim0_stride[31:0]
  g1[6] = 0u;                                               // stride hi, tensor_dim1_stride lo
  g1[7] = 0u;
  asm volatile("tensor_load_to_lds %0, %1" :: "s"(g0), "s"(g1) : "memory");
}

// ---------------- prep kernels ----------------

__global__ void cvt_f16_kernel(const float* __restrict__ s, _Float16* __restrict__ d, int n) {
  int i = blockIdx.x * blockDim.x + threadIdx.x;
  if (i < n) d[i] = (_Float16)s[i];
}

__global__ void zero_f16_kernel(_Float16* __restrict__ d, int n) {
  int i = blockIdx.x * blockDim.x + threadIdx.x;
  if (i < n) d[i] = (_Float16)0.0f;
}

__global__ void gather_tgt_kernel(const int* __restrict__ tg, const float* __restrict__ emb,
                                  _Float16* __restrict__ d) {
  int i = blockIdx.x * blockDim.x + threadIdx.x;   // (25*24) x 128
  if (i >= kY1 * kC * kD) return;
  int r = i >> 7, k = i & 127;
  d[i] = (_Float16)emb[(size_t)tg[r] * kD + k];
}

__global__ void build_heads_kernel(const float* __restrict__ Ws, const float* __restrict__ bs,
                                   const float* __restrict__ Wi, const float* __restrict__ bi,
                                   _Float16* __restrict__ Wh, float* __restrict__ bh) {
  int i = blockIdx.x * blockDim.x + threadIdx.x;   // 128 x 256
  if (i >= kNHead * kH) return;
  int n = i >> 8, k = i & 255;
  float v = 0.0f;
  if (n < 62) v = Ws[n * kH + k];
  else if (n >= 64 && n < 126) v = Wi[(n - 64) * kH + k];
  Wh[i] = (_Float16)v;
  if (k == 0) {
    float b = 0.0f;
    if (n < 62) b = bs[n];
    else if (n >= 64 && n < 126) b = bi[n - 64];
    bh[n] = b;
  }
}

// ---------------- generic batched WMMA GEMM:  C = remap(A) @ W^T + b0 + b1 ----------------
// A: f16, rows mapped (t, m) with optional time reversal; pad rows read a zeroed dummy row
// (branch-free inner loop). W: f16 (N x K); C: f32. One wave per 16x16 output tile.

__global__ __launch_bounds__(256) void gemm_f16_kernel(
    const _Float16* __restrict__ A, const _Float16* __restrict__ W,
    const float* __restrict__ b0, const float* __restrict__ b1, float* __restrict__ C,
    const _Float16* __restrict__ zrow,
    int Mtiles, int Ntiles, int K, int rows_out, int rows_src, int T, int rev) {
  const int lane = threadIdx.x & 31;
  const int wid = threadIdx.x >> 5;
  int tile = blockIdx.x * 8 + wid;
  if (tile >= Mtiles * Ntiles) return;
  const int mt = tile / Ntiles, nt = tile % Ntiles;
  const int r = lane & 15, hi = lane >> 4;
  const int hi8 = hi * 8, kh = hi * 16;
  const int N = Ntiles * 16;

  // A row remap: gr -> (t, mm); pad rows (mm >= rows_src) read the zero row
  int gr = mt * 16 + r;
  int t = gr / rows_out;
  int mm = gr - t * rows_out;
  const _Float16* arow = zrow;
  if (mm < rows_src && t < T) {
    int st = rev ? (T - 1 - t) : t;
    arow = A + (size_t)(st * rows_src + mm) * K;
  }

  const int gn = nt * 16 + r;
  float binit = 0.0f;
  if (b0) binit += b0[gn];
  if (b1) binit += b1[gn];
  v8f acc;
#pragma unroll
  for (int v = 0; v < 8; ++v) acc[v] = binit;

  for (int kc = 0; kc < K; kc += 32) {
    v16h a = ldA(arow, kc, hi8);
    v16h b = ldB(W, gn, K, kc, kh);
    acc = wmma_f16(a, b, acc);
  }

  float* crow = C + (size_t)(mt * 16) * N + nt * 16 + r;
#pragma unroll
  for (int v = 0; v < 8; ++v) crow[(size_t)(v + 8 * hi) * N] = acc[v];
}

// ---------------- LSTM scan: one block owns a 16-row batch slice for all T steps ----------
// z[t] = zpre[t] + h @ Whh^T (WMMA). For HD==128 the contiguous 32KB zpre[t] tile is
// staged into one LDS buffer by the Tensor Data Mover (issued for t+1 during step t's
// elementwise phase, waited with s_wait_tensorcnt). Gates in LDS, cell state in registers.

template <int HD>
__global__ __launch_bounds__(256) void lstm_scan_kernel(
    const float* __restrict__ zpreA, const _Float16* __restrict__ WhhA, float* __restrict__ houtA,
    const float* __restrict__ zpreB, const _Float16* __restrict__ WhhB, float* __restrict__ houtB,
    int nSlices, int Mtot, int T) {
  constexpr int N4 = 4 * HD;
  constexpr int NT = N4 / 16;    // n-tiles
  constexpr int TPW = NT / 8;    // tiles per wave (8 waves / block)
  constexpr int CPT = HD / 16;   // cell elems per thread
  constexpr bool kTdm = (HD == 128);   // 16*N4*4 = 32KB stage fits the LDS budget
  __shared__ _Float16 zsh[16 * N4];
  __shared__ _Float16 hsh[16 * HD];
  __shared__ float zstage[kTdm ? 16 * N4 : 1];

  const int lane = threadIdx.x & 31;
  const int wid = threadIdx.x >> 5;
  const int dir = blockIdx.x / nSlices;
  const int slice = blockIdx.x % nSlices;
  const int gm0 = slice * 16;
  const float* zpre = dir ? zpreB : zpreA;
  const _Float16* Whh = dir ? WhhB : WhhA;
  float* hout = dir ? houtB : houtA;

  const int r = lane & 15, hi = lane >> 4;
  const int hi8 = hi * 8, kh = hi * 16;
  const int em = threadIdx.x >> 4;   // 0..15 row for elementwise
  const int ek = threadIdx.x & 15;

  __builtin_prefetch(Whh + (size_t)threadIdx.x * 64, 0, 1);  // warm recurrent weights (L2)

  float creg[CPT];
#pragma unroll
  for (int j = 0; j < CPT; ++j) {
    creg[j] = 0.0f;
    hsh[em * HD + ek + 16 * j] = (_Float16)0.0f;
  }
  if constexpr (kTdm) {
    if (wid == 0) {   // stage zpre[0]; tensor ops ignore EXEC, one wave issues the DMA
      tdm_load_1d_f32(zpre + (size_t)gm0 * N4, (unsigned)(uintptr_t)&zstage[0], 16 * N4);
      __builtin_amdgcn_s_wait_tensorcnt(0);
    }
  }
  __syncthreads();

  for (int t = 0; t < T; ++t) {
    // --- recurrent GEMM: z = zpre + h @ Whh^T ---
#pragma unroll
    for (int jt = 0; jt < TPW; ++jt) {
      const int n0 = (wid * TPW + jt) * 16;
      const int gn = n0 + r;
      v8f acc;
      if constexpr (kTdm) {
        const float* zp = &zstage[gn];
#pragma unroll
        for (int v = 0; v < 8; ++v) acc[v] = zp[(v + 8 * hi) * N4];
      } else {
        const float* zp = zpre + ((size_t)t * Mtot + gm0) * N4 + gn;
#pragma unroll
        for (int v = 0; v < 8; ++v) acc[v] = zp[(size_t)(v + 8 * hi) * N4];
      }
#pragma unroll
      for (int kc = 0; kc < HD; kc += 32) {
        v16h a = ldA(&hsh[r * HD], kc, hi8);           // LDS A fragment
        v16h b = ldB(Whh, gn, HD, kc, kh);             // global B fragment (L2-hot)
        acc = wmma_f16(a, b, acc);
      }
#pragma unroll
      for (int v = 0; v < 8; ++v) zsh[(v + 8 * hi) * N4 + n0 + r] = (_Float16)acc[v];
    }
    __syncthreads();                                    // zstage fully consumed past this point
    if constexpr (kTdm) {
      if (wid == 0 && t + 1 < T)                        // overlap next tile DMA with elementwise
        tdm_load_1d_f32(zpre + ((size_t)(t + 1) * Mtot + gm0) * N4,
                        (unsigned)(uintptr_t)&zstage[0], 16 * N4);
    }
    // --- gates / cell / hidden ---
#pragma unroll
    for (int j = 0; j < CPT; ++j) {
      const int k = ek + 16 * j;
      float zi = (float)zsh[em * N4 + k];
      float zf = (float)zsh[em * N4 + HD + k];
      float zg = (float)zsh[em * N4 + 2 * HD + k];
      float zo = (float)zsh[em * N4 + 3 * HD + k];
      float c = sigm(zf) * creg[j] + sigm(zi) * tanhf(zg);
      creg[j] = c;
      float h = sigm(zo) * tanhf(c);
      hout[((size_t)t * Mtot + gm0 + em) * HD + k] = h;
      hsh[em * HD + k] = (_Float16)h;
    }
    if constexpr (kTdm) {
      if (wid == 0) __builtin_amdgcn_s_wait_tensorcnt(0);
    }
    __syncthreads();
  }
}

// ---------------- pack prior/modern hidden states to f16 GEMM inputs ----------------

__global__ void build_ctx_kernel(const float* __restrict__ hf, const float* __restrict__ hb,
                                 const float* __restrict__ hm, _Float16* __restrict__ priorh,
                                 _Float16* __restrict__ modernh) {
  int i = blockIdx.x * blockDim.x + threadIdx.x;
  const int NP = kX1 * kCB * kH;  // 4800 x 256
  if (i < NP) {
    int rr = i >> 8, k = i & 255;
    int t = rr / kCB, m = rr % kCB;
    float v;
    if (k < kHD) v = hf[((size_t)t * kCB + m) * kHD + k];
    else         v = hb[((size_t)(kX2 - 1 - t) * kCB + m) * kHD + (k - kHD)];
    priorh[i] = (_Float16)v;
  } else {
    int j = i - NP;                // 600 x 256
    if (j >= kY1 * kC * kH) return;
    int rr = j >> 8, k = j & 255;
    int t = rr / kC, cc = rr % kC;
    modernh[j] = (_Float16)hm[((size_t)t * 32 + cc) * kH + k];
  }
}

// ---------------- fused broadcast-add + dual log-softmax, one wave per row ----------------

__global__ __launch_bounds__(256) void fuse_logsoftmax_kernel(const float* __restrict__ plog,
                                                              const float* __restrict__ mlog,
                                                              float* __restrict__ out) {
  const int lane = threadIdx.x & 31;
  const int wid = threadIdx.x >> 5;
  long R = (long)blockIdx.x * 8 + wid;
  if (R >= kRows) return;
  int x = (int)(R / 4800);
  int rem = (int)(R % 4800);
  int y = rem / 192;
  int m = rem % 192;          // = c*8 + b
  int c = m >> 3;

  const float* pr = plog + ((size_t)x * kCB + m) * kNHead;
  const float* mr = mlog + ((size_t)y * kC + c) * kNHead;
  float v0 = pr[lane] + mr[lane];
  float v1 = pr[lane + 32] + mr[lane + 32];
  float v2 = pr[lane + 64] + mr[lane + 64];
  float v3 = pr[lane + 96] + mr[lane + 96];
  const bool hiv = lane < 30;   // o = lane+32 < 62

  size_t base = (size_t)R * 62;
  auto dohead = [&](float a, float b, float* ob) {
    float mx = fmaxf(a, hiv ? b : -__builtin_inff());
#pragma unroll
    for (int s = 16; s > 0; s >>= 1) mx = fmaxf(mx, __shfl_xor(mx, s, 32));
    float sum = __expf(a - mx) + (hiv ? __expf(b - mx) : 0.0f);
#pragma unroll
    for (int s = 16; s > 0; s >>= 1) sum += __shfl_xor(sum, s, 32);
    float lse = mx + __logf(sum);
    ob[lane] = a - lse;
    if (hiv) ob[lane + 32] = b - lse;
  };
  dohead(v0, v1, out + base);                 // sub head (cols 0..61)
  dohead(v2, v3, out + kOutHead + base);      // ins head (cols 64..125)
}

// ---------------- host launcher ----------------

extern "C" void kernel_launch(void* const* d_in, const int* in_sizes, int n_in,
                              void* d_out, int out_size, void* d_ws, size_t ws_size,
                              hipStream_t stream) {
  (void)in_sizes; (void)n_in; (void)out_size; (void)ws_size;
  const float* src_emb = (const float*)d_in[0];
  const int*   targets = (const int*)d_in[1];
  const float* emb     = (const float*)d_in[2];
  const float* Wih_f = (const float*)d_in[3];
  const float* Whh_f = (const float*)d_in[4];
  const float* bih_f = (const float*)d_in[5];
  const float* bhh_f = (const float*)d_in[6];
  const float* Wih_b = (const float*)d_in[7];
  const float* Whh_b = (const float*)d_in[8];
  const float* bih_b = (const float*)d_in[9];
  const float* bhh_b = (const float*)d_in[10];
  const float* Wih_m = (const float*)d_in[11];
  const float* Whh_m = (const float*)d_in[12];
  const float* bih_m = (const float*)d_in[13];
  const float* bhh_m = (const float*)d_in[14];
  const float* W_sub = (const float*)d_in[15];
  const float* b_sub = (const float*)d_in[16];
  const float* W_ins = (const float*)d_in[17];
  const float* b_ins = (const float*)d_in[18];
  float* out = (float*)d_out;

  char* ws = (char*)d_ws;
  size_t off = 0;
  auto alloc = [&](size_t bytes) -> char* {
    char* p = ws + off;
    off += (bytes + 255) & ~(size_t)255;
    return p;
  };

  _Float16* Asrc   = (_Float16*)alloc((size_t)kX2 * kCB * kD * 2);     // 4992 x 128
  _Float16* Wihf16 = (_Float16*)alloc(512 * 128 * 2);
  _Float16* Whhf16 = (_Float16*)alloc(512 * 128 * 2);
  _Float16* Wihb16 = (_Float16*)alloc(512 * 128 * 2);
  _Float16* Whhb16 = (_Float16*)alloc(512 * 128 * 2);
  _Float16* Wihm16 = (_Float16*)alloc(1024 * 128 * 2);
  _Float16* Whhm16 = (_Float16*)alloc(1024 * 256 * 2);
  _Float16* Whead  = (_Float16*)alloc((size_t)kNHead * kH * 2);
  float*    bhead  = (float*)alloc(kNHead * 4);
  _Float16* Tgt16  = (_Float16*)alloc((size_t)kY1 * kC * kD * 2);      // 600 x 128
  _Float16* zrow   = (_Float16*)alloc(256 * 2);                        // zero pad row (K<=256)
  float* zpreF = (float*)alloc((size_t)kX2 * kCB * 512 * 4);
  float* zpreB = (float*)alloc((size_t)kX2 * kCB * 512 * 4);
  float* zpreM = (float*)alloc((size_t)kY1 * 32 * 1024 * 4);
  float* hF = (float*)alloc((size_t)kX2 * kCB * kHD * 4);
  float* hB = (float*)alloc((size_t)kX2 * kCB * kHD * 4);
  float* hM = (float*)alloc((size_t)kY1 * 32 * kH * 4);
  _Float16* priorh  = (_Float16*)alloc((size_t)kX1 * kCB * kH * 2);    // 4800 x 256
  _Float16* modernh = (_Float16*)alloc((size_t)kY1 * kC * kH * 2);     // 600 x 256
  float* plog = (float*)alloc((size_t)kX1 * kCB * kNHead * 4);         // 4800 x 128
  float* mlog = (float*)alloc((size_t)608 * kNHead * 4);               // 608 x 128 (padded)

  auto cvt = [&](const float* s, _Float16* d, int n) {
    cvt_f16_kernel<<<(n + 255) / 256, 256, 0, stream>>>(s, d, n);
  };
  cvt(src_emb, Asrc, kX2 * kCB * kD);
  cvt(Wih_f, Wihf16, 512 * 128);
  cvt(Whh_f, Whhf16, 512 * 128);
  cvt(Wih_b, Wihb16, 512 * 128);
  cvt(Whh_b, Whhb16, 512 * 128);
  cvt(Wih_m, Wihm16, 1024 * 128);
  cvt(Whh_m, Whhm16, 1024 * 256);
  zero_f16_kernel<<<1, 256, 0, stream>>>(zrow, 256);
  gather_tgt_kernel<<<(kY1 * kC * kD + 255) / 256, 256, 0, stream>>>(targets, emb, Tgt16);
  build_heads_kernel<<<(kNHead * kH + 255) / 256, 256, 0, stream>>>(W_sub, b_sub, W_ins, b_ins,
                                                                    Whead, bhead);

  // Batched input projections across all timesteps (parallel-in-T): z_pre = x @ Wih^T + bih + bhh
  gemm_f16_kernel<<<1248, 256, 0, stream>>>(Asrc, Wihf16, bih_f, bhh_f, zpreF, zrow,
                                            312, 32, 128, 192, 192, kX2, 0);
  gemm_f16_kernel<<<1248, 256, 0, stream>>>(Asrc, Wihb16, bih_b, bhh_b, zpreB, zrow,
                                            312, 32, 128, 192, 192, kX2, 1);   // time-reversed
  gemm_f16_kernel<<<400, 256, 0, stream>>>(Tgt16, Wihm16, bih_m, bhh_m, zpreM, zrow,
                                           50, 64, 128, 32, 24, kY1, 0);       // batch pad 24->32

  // Recurrent scans: 24 blocks (12 fwd slices + 12 bwd slices), 2 blocks for modern.
  lstm_scan_kernel<128><<<24, 256, 0, stream>>>(zpreF, Whhf16, hF, zpreB, Whhb16, hB, 12, 192, kX2);
  lstm_scan_kernel<256><<<2, 256, 0, stream>>>(zpreM, Whhm16, hM, zpreM, Whhm16, hM, 2, 32, kY1);

  // Pack prior (fwd||bwd) and modern to f16 for the factored head GEMMs.
  {
    int n = kX1 * kCB * kH + kY1 * kC * kH;
    build_ctx_kernel<<<(n + 255) / 256, 256, 0, stream>>>(hF, hB, hM, priorh, modernh);
  }

  // Factored heads: logits(ctx) = prior@W^T (+bias) + modern@W^T ; both heads packed into N=128.
  gemm_f16_kernel<<<300, 256, 0, stream>>>(priorh, Whead, bhead, nullptr, plog, zrow,
                                           300, 8, 256, 4800, 4800, 1, 0);
  gemm_f16_kernel<<<38, 256, 0, stream>>>(modernh, Whead, nullptr, nullptr, mlog, zrow,
                                          38, 8, 256, 608, 600, 1, 0);

  // Fused broadcast add + dual log-softmax, one wave per output row.
  fuse_logsoftmax_kernel<<<(int)((kRows + 7) / 8), 256, 0, stream>>>(plog, mlog, out);
}
